// TransformerBlock_9199819948123
// MI455X (gfx1250) — matrix operand order repaired and checked by reading, not compiled, not run on hardware
//
#include <hip/hip_runtime.h>
#include <hip/hip_bf16.h>

typedef __attribute__((ext_vector_type(8)))  _Float16 v8h;
typedef __attribute__((ext_vector_type(16))) _Float16 v16h;
typedef __attribute__((ext_vector_type(8)))  float    v8f;
typedef __attribute__((ext_vector_type(4)))  unsigned int u32x4;
typedef __attribute__((ext_vector_type(4)))  int      i32x4;
typedef __attribute__((ext_vector_type(8)))  int      i32x8;

#define TB_B   2
#define TB_S   2048
#define TB_D   1024
#define TB_H   16
#define TB_DH  64
#define TB_DM  4096
#define TB_TOK (TB_B * TB_S)   // 4096 rows

#if __has_builtin(__builtin_amdgcn_tensor_load_to_lds)
#define HAVE_TDM 1
#else
#define HAVE_TDM 0
#endif

// ---------------------------------------------------------------------------
// CDNA5 async global->LDS copy (ASYNCcnt): no VGPR round-trip.
// ---------------------------------------------------------------------------
__device__ __forceinline__ void async_copy_b128(void* lds_dst, const void* gsrc) {
  unsigned int l = (unsigned int)(uintptr_t)lds_dst;  // flat low 32b == LDS addr
  asm volatile("global_load_async_to_lds_b128 %0, %1, off"
               :: "v"(l), "v"(gsrc) : "memory");
}
__device__ __forceinline__ void wait_async() {
  asm volatile("s_wait_asynccnt 0x0" ::: "memory");
}

#if HAVE_TDM
// ---------------------------------------------------------------------------
// Tensor Data Mover: 2D tile (nrows x 64) f16, global row stride
// stride_elems, LDS rows padded 128B + 16B (pad_interval=4 -> 16 x 8B units,
// pad_amount=3 -> 4 DWORDs) to match LDS leading dim of 72 halfs.
// D# layout per CDNA5 ISA.
// ---------------------------------------------------------------------------
__device__ __forceinline__ void tdm_load_tile_f16(unsigned int lds_addr,
                                                  const void* gptr,
                                                  unsigned int nrows,
                                                  unsigned int stride_elems) {
  unsigned long long ga = (unsigned long long)(uintptr_t)gptr;
  u32x4 g0;
  g0[0] = 1u;                                   // count=1, user mode
  g0[1] = lds_addr;                             // lds_addr [63:32]
  g0[2] = (unsigned int)(ga & 0xffffffffu);     // global_addr [95:64]
  g0[3] = (unsigned int)((ga >> 32) & 0x01ffffffu) | (2u << 30); // type=2
  i32x8 g1;
  g1[0] = (int)((1u << 16) | (1u << 20) | (4u << 22) | (3u << 25));
          // data_size=2B, pad_enable, pad_interval=16x8B, pad_amount=4 dw
  g1[1] = (int)(64u << 16);                     // tensor_dim0 = 64
  g1[2] = (int)((nrows & 0xffffu) << 16);       // tensor_dim1 = nrows
  g1[3] = (int)(64u << 16);                     // tile_dim0   = 64
  g1[4] = (int)nrows;                           // tile_dim1   = nrows
  g1[5] = (int)stride_elems;                    // tensor_dim0_stride
  g1[6] = 0;
  g1[7] = 0;
  i32x4 z4 = {0, 0, 0, 0};
#if defined(__clang_major__) && (__clang_major__ >= 23)
  i32x8 z8 = {0, 0, 0, 0, 0, 0, 0, 0};
  __builtin_amdgcn_tensor_load_to_lds(g0, g1, z4, z4, z8, 0);
#else
  __builtin_amdgcn_tensor_load_to_lds(g0, g1, z4, z4, 0);
#endif
}
#endif

// ---------------------------------------------------------------------------
// WMMA helpers (CDNA5 16x16x32 f16 -> f32 accumulate), contiguous k
// order (k = 16*(lane/16) + s) for both A and B, not the ISA table order.
// ---------------------------------------------------------------------------
__device__ __forceinline__ v8f wmma_f16(v16h a, v16h b, v8f c) {
  return __builtin_amdgcn_wmma_f32_16x16x32_f16(false, a, false, b,
                                                (short)0, c, false, false);
}

__device__ __forceinline__ v16h load_a_frag(const _Float16* p) {
  v8h lo = *(const v8h*)(p);        // K = kbase + 0..7
  v8h hi = *(const v8h*)(p + 8);    // K = kbase + 8..15
  v16h r;
#pragma unroll
  for (int i = 0; i < 8; ++i) { r[i] = lo[i]; r[i + 8] = hi[i]; }
  return r;
}

__device__ __forceinline__ v16h load_b_frag(const _Float16* p) {
  v8h lo = *(const v8h*)(p);        // 16 contiguous K values
  v8h hi = *(const v8h*)(p + 8);
  v16h r;
#pragma unroll
  for (int i = 0; i < 8; ++i) { r[i] = lo[i]; r[i + 8] = hi[i]; }
  return r;
}

__device__ __forceinline__ float gelu_tanh(float x) {
  float x3 = x * x * x;
  return 0.5f * x * (1.0f + tanhf(0.7978845608028654f * (x + 0.044715f * x3)));
}

// ---------------------------------------------------------------------------
// LayerNorm (TransformerLens style) + fp32 -> fp16 cast. One block per row.
// ---------------------------------------------------------------------------
__global__ __launch_bounds__(256) void layernorm_f16_kernel(
    const float* __restrict__ x, const float* __restrict__ w,
    const float* __restrict__ b, _Float16* __restrict__ y) {
  __shared__ float red[256];
  const int row = blockIdx.x;
  const int t = threadIdx.x;
  const float* xr = x + (size_t)row * TB_D;

  float v[4];
  float s = 0.f;
#pragma unroll
  for (int i = 0; i < 4; ++i) { v[i] = xr[t + i * 256]; s += v[i]; }
  red[t] = s;
  __syncthreads();
  for (int off = 128; off > 0; off >>= 1) {
    if (t < off) red[t] += red[t + off];
    __syncthreads();
  }
  const float mean = red[0] * (1.0f / TB_D);
  __syncthreads();

  float ss = 0.f;
#pragma unroll
  for (int i = 0; i < 4; ++i) { v[i] -= mean; ss += v[i] * v[i]; }
  red[t] = ss;
  __syncthreads();
  for (int off = 128; off > 0; off >>= 1) {
    if (t < off) red[t] += red[t + off];
    __syncthreads();
  }
  const float inv = rsqrtf(red[0] * (1.0f / TB_D) + 1e-5f);

#pragma unroll
  for (int i = 0; i < 4; ++i) {
    const int c = t + i * 256;
    y[(size_t)row * TB_D + c] = (_Float16)(v[i] * inv * w[c] + b[c]);
  }
}

// ---------------------------------------------------------------------------
// Generic WMMA GEMM:  C[M,N] (+bias)(gelu?)(+resid?) = A[M,K](f16) x B[K,N](f32)
// Block 128x64, K-step 64, 8 waves x (16 rows x 64 cols).
// Double-buffered (ping-pong) main loop: next A tile staged by the Tensor
// Data Mover (TENSORcnt) or per-lane async global->LDS (ASYNCcnt), next B
// tile's global loads issued before the current 8-WMMA chain; waits only
// after compute, so staging latency is off the critical path.
// ---------------------------------------------------------------------------
#define G_LDA 72
#define G_LDB 72

__global__ __launch_bounds__(256) void wmma_gemm_kernel(
    const _Float16* __restrict__ A, int lda,
    const float* __restrict__ B, int ldb, long long strideBz,
    const float* __restrict__ bias, int biasStrideZ,
    const float* __restrict__ resid,
    _Float16* __restrict__ C16, float* __restrict__ C32,
    int ldc, int cColStrideZ, int K, int doGelu) {
  __shared__ _Float16 As[2][128 * G_LDA];
  __shared__ _Float16 Bs[2][64 * G_LDB];

  const int t = threadIdx.x;
  const int lane = t & 31, wid = t >> 5;
  const int rowBase = blockIdx.x * 128;
  const int colLocalBase = blockIdx.y * 64;
  const int z = blockIdx.z;
  const float* Bz = B + (long long)z * strideBz;

  v8f acc[4];
#pragma unroll
  for (int nt = 0; nt < 4; ++nt)
#pragma unroll
    for (int i = 0; i < 8; ++i) acc[nt][i] = 0.f;

  const int m0 = wid * 16;
  const int ml = lane & 15;
  const int kb8 = (lane >> 4) * 8;
  const int kb16 = (lane >> 4) * 16;

  // ---- tile staging helpers --------------------------------------------
  auto stage_A = [&](int buf, int k0) {
#if HAVE_TDM
    if (wid == 0)
      tdm_load_tile_f16((unsigned int)(uintptr_t)As[buf],
                        &A[(size_t)rowBase * lda + k0], 128,
                        (unsigned int)lda);
#else
#pragma unroll
    for (int i = 0; i < 4; ++i) {
      const int c = t + i * 256;           // 1024 chunks of 8 halfs
      const int r = c >> 3, ch = (c & 7) * 8;
      async_copy_b128(&As[buf][r * G_LDA + ch],
                      &A[(size_t)(rowBase + r) * lda + k0 + ch]);
    }
#endif
  };
  auto stage_B = [&](int buf, int k0) {
#pragma unroll
    for (int i = 0; i < 16; ++i) {
      const int e = t + i * 256;
      const int r = e >> 6, cc = e & 63;
      Bs[buf][cc * G_LDB + r] =
          (_Float16)Bz[(size_t)(k0 + r) * ldb + colLocalBase + cc];
    }
  };
  auto stage_wait = [&]() {
#if HAVE_TDM
    if (wid == 0) __builtin_amdgcn_s_wait_tensorcnt(0);
#else
    wait_async();
#endif
  };

  // ---- prologue: stage tile 0 ------------------------------------------
  stage_A(0, 0);
  stage_B(0, 0);
  stage_wait();
  __syncthreads();

  int p = 0;
  for (int k0 = 0; k0 < K; k0 += 64) {
    const bool hasNext = (k0 + 64) < K;
    if (hasNext) {             // kick off next tile before computing
      stage_A(1 - p, k0 + 64);
      stage_B(1 - p, k0 + 64);
    }

    // ---- load all fragments, then back-to-back WMMA ---------------------
    const _Float16* ap = &As[p][(m0 + ml) * G_LDA + kb16];
    const v16h a0 = load_a_frag(ap);
    const v16h a1 = load_a_frag(ap + 32);
    v16h bf[8];
#pragma unroll
    for (int nt = 0; nt < 4; ++nt) {
      const _Float16* bp = &Bs[p][(nt * 16 + ml) * G_LDB + kb16];
      bf[nt * 2]     = load_b_frag(bp);
      bf[nt * 2 + 1] = load_b_frag(bp + 32);
    }
#pragma unroll
    for (int nt = 0; nt < 4; ++nt) {
      acc[nt] = wmma_f16(a0, bf[nt * 2],     acc[nt]);
      acc[nt] = wmma_f16(a1, bf[nt * 2 + 1], acc[nt]);
    }

    if (hasNext) stage_wait();
    __syncthreads();           // compute done on p; next tile landed in 1-p
    p ^= 1;
  }

  // ---- epilogue ---------------------------------------------------------
  const int rhalf = (lane >> 4) * 8;
#pragma unroll
  for (int nt = 0; nt < 4; ++nt) {
    const int colLocal = colLocalBase + nt * 16 + ml;
    const int col = z * cColStrideZ + colLocal;
    const float bv = bias ? bias[(size_t)z * biasStrideZ + colLocal] : 0.f;
#pragma unroll
    for (int v = 0; v < 8; ++v) {
      const int row = rowBase + m0 + rhalf + v;
      float x = acc[nt][v] + bv;
      if (doGelu) x = gelu_tanh(x);
      const size_t idx = (size_t)row * ldc + col;
      if (C16) C16[idx] = (_Float16)x;
      else     C32[idx] = x + (resid ? resid[idx] : 0.f);
    }
  }
}

// ---------------------------------------------------------------------------
// Causal flash attention. Grid (S/64, H, B), 128 threads (4 waves).
// Q staged via async LDS copies; K tile via the Tensor Data Mover (wave 0
// issues the D#, waits TENSORcnt after V staging); V transposed in VGPRs.
// ---------------------------------------------------------------------------
#define A_LD 72

__global__ __launch_bounds__(128) void attn_kernel(
    const _Float16* __restrict__ qb, const _Float16* __restrict__ kbuf,
    const _Float16* __restrict__ vb, _Float16* __restrict__ zb) {
  __shared__ _Float16 Qs[64 * A_LD];   // [q][dim]
  __shared__ _Float16 Ks[64 * A_LD];   // [key][dim]  == [N][K] for QK^T
  __shared__ _Float16 Vt[64 * A_LD];   // [dim][key]  == [N][K] for P*V
  __shared__ _Float16 Ps[64 * A_LD];   // probs [q][key]
  __shared__ float    Sc[64 * 65];     // raw scores f32
  __shared__ float    mrow[64], lrow[64], srow[64];

  const int t = threadIdx.x;
  const int lane = t & 31, wid = t >> 5;
  const int qblk = blockIdx.x, h = blockIdx.y, b = blockIdx.z;
  const int qstart = qblk * 64;
  const size_t base = (size_t)b * TB_S * (TB_H * TB_DH) + h * TB_DH;

  // load Q tile once (async)
#pragma unroll
  for (int i = 0; i < 4; ++i) {
    const int c = t + i * 128;
    const int r = c >> 3, c8 = (c & 7) * 8;
    async_copy_b128(&Qs[r * A_LD + c8],
                    &qb[base + (size_t)(qstart + r) * (TB_H * TB_DH) + c8]);
  }
  if (t < 64) { mrow[t] = -3.0e38f; lrow[t] = 0.f; }

  v8f oacc[4];
#pragma unroll
  for (int nt = 0; nt < 4; ++nt)
#pragma unroll
    for (int i = 0; i < 8; ++i) oacc[nt][i] = 0.f;
  wait_async();
  __syncthreads();

  const int m0 = wid * 16;
  const int ml = lane & 15;
  const int kb8 = (lane >> 4) * 8;
  const int kb16 = (lane >> 4) * 16;

  for (int kblk = 0; kblk <= qblk; ++kblk) {
    // ---- stage K tile (TDM issues, waits later) -------------------------
#if HAVE_TDM
    if (wid == 0)
      tdm_load_tile_f16((unsigned int)(uintptr_t)Ks,
                        &kbuf[base + (size_t)(kblk * 64) * (TB_H * TB_DH)],
                        64, TB_H * TB_DH);
#else
#pragma unroll
    for (int i = 0; i < 4; ++i) {
      const int c = t + i * 128;
      const int r = c >> 3, c8 = (c & 7) * 8;
      async_copy_b128(&Ks[r * A_LD + c8],
                      &kbuf[base + (size_t)(kblk * 64 + r) * (TB_H * TB_DH) + c8]);
    }
#endif
    // ---- stage V transposed [dim][key] ----------------------------------
#pragma unroll
    for (int i = 0; i < 4; ++i) {
      const int c = t + i * 128;
      const int r = c >> 3, c8 = (c & 7) * 8;
      const v8h vv =
          *(const v8h*)&vb[base + (size_t)(kblk * 64 + r) * (TB_H * TB_DH) + c8];
#pragma unroll
      for (int j = 0; j < 8; ++j) Vt[(c8 + j) * A_LD + r] = vv[j];
    }
#if HAVE_TDM
    if (wid == 0) __builtin_amdgcn_s_wait_tensorcnt(0);
#else
    wait_async();
#endif
    __syncthreads();

    // ---- scores = Q x K^T / sqrt(DH) ------------------------------------
    {
      const _Float16* ap = &Qs[(m0 + ml) * A_LD + kb16];
      const v16h a0 = load_a_frag(ap);
      const v16h a1 = load_a_frag(ap + 32);
#pragma unroll
      for (int nt = 0; nt < 4; ++nt) {
        const _Float16* bp = &Ks[(nt * 16 + ml) * A_LD + kb16];
        const v16h b0 = load_b_frag(bp);
        const v16h b1 = load_b_frag(bp + 32);
        v8f s;
#pragma unroll
        for (int i = 0; i < 8; ++i) s[i] = 0.f;
        s = wmma_f16(a0, b0, s);
        s = wmma_f16(a1, b1, s);
        const int col = nt * 16 + ml;
#pragma unroll
        for (int v = 0; v < 8; ++v)
          Sc[(m0 + (lane >> 4) * 8 + v) * 65 + col] = s[v] * 0.125f;
      }
    }
    __syncthreads();

    // ---- online softmax (one thread per q row) --------------------------
    if (t < 64) {
      const int r = t;
      const int nvalid = (kblk == qblk) ? (r + 1) : 64;  // causal mask
      const float mo = mrow[r];
      float mx = mo;
      for (int j = 0; j < nvalid; ++j) mx = fmaxf(mx, Sc[r * 65 + j]);
      float sum = 0.f;
      for (int j = 0; j < 64; ++j) {
        const float p = (j < nvalid) ? __expf(Sc[r * 65 + j] - mx) : 0.f;
        Ps[r * A_LD + j] = (_Float16)p;
        sum += p;
      }
      const float sc = __expf(mo - mx);
      mrow[r] = mx;
      lrow[r] = lrow[r] * sc + sum;
      srow[r] = sc;
    }
    __syncthreads();

    // ---- rescale running output, then O += P x V ------------------------
#pragma unroll
    for (int nt = 0; nt < 4; ++nt)
#pragma unroll
      for (int v = 0; v < 8; ++v)
        oacc[nt][v] *= srow[m0 + (lane >> 4) * 8 + v];
    {
      const _Float16* ap = &Ps[(m0 + ml) * A_LD + kb16];
      const v16h a0 = load_a_frag(ap);
      const v16h a1 = load_a_frag(ap + 32);
#pragma unroll
      for (int nt = 0; nt < 4; ++nt) {
        const _Float16* bp = &Vt[(nt * 16 + ml) * A_LD + kb16];
        const v16h b0 = load_b_frag(bp);
        const v16h b1 = load_b_frag(bp + 32);
        oacc[nt] = wmma_f16(a0, b0, oacc[nt]);
        oacc[nt] = wmma_f16(a1, b1, oacc[nt]);
      }
    }
    __syncthreads();
  }

  // ---- normalize and store z (f16, [TOK, H*DH]) -------------------------
#pragma unroll
  for (int nt = 0; nt < 4; ++nt) {
#pragma unroll
    for (int v = 0; v < 8; ++v) {
      const int r = m0 + (lane >> 4) * 8 + v;
      const float invl = 1.0f / lrow[r];
      zb[base + (size_t)(qstart + r) * (TB_H * TB_DH) + nt * 16 + ml] =
          (_Float16)(oacc[nt][v] * invl);
    }
  }
}

// ---------------------------------------------------------------------------
// Host-side launch sequence (graph-capture safe: only kernel launches).
// ---------------------------------------------------------------------------
extern "C" void kernel_launch(void* const* d_in, const int* in_sizes, int n_in,
                              void* d_out, int out_size, void* d_ws,
                              size_t ws_size, hipStream_t stream) {
  const float* resid_pre = (const float*)d_in[0];
  const float* ln1_w = (const float*)d_in[1];
  const float* ln1_b = (const float*)d_in[2];
  const float* W_Q   = (const float*)d_in[3];
  const float* W_K   = (const float*)d_in[4];
  const float* W_V   = (const float*)d_in[5];
  const float* W_O   = (const float*)d_in[6];
  const float* b_Q   = (const float*)d_in[7];
  const float* b_K   = (const float*)d_in[8];
  const float* b_V   = (const float*)d_in[9];
  const float* b_O   = (const float*)d_in[10];
  const float* ln2_w = (const float*)d_in[11];
  const float* ln2_b = (const float*)d_in[12];
  const float* W_in  = (const float*)d_in[13];
  const float* b_in  = (const float*)d_in[14];
  const float* W_out = (const float*)d_in[15];
  const float* b_out = (const float*)d_in[16];
  float* out = (float*)d_out;          // resid_mid lives here, then final out

  char* ws = (char*)d_ws;
  const size_t HB = (size_t)TB_TOK * TB_D * sizeof(_Float16);  // 8 MB
  _Float16* xln = (_Float16*)(ws);
  _Float16* qb  = (_Float16*)(ws + 1 * HB);
  _Float16* kb  = (_Float16*)(ws + 2 * HB);
  _Float16* vb  = (_Float16*)(ws + 3 * HB);
  _Float16* zb  = (_Float16*)(ws + 4 * HB);
  _Float16* hid = (_Float16*)(ws + 5 * HB);    // 32 MB: [TOK, DM]

  const long long wqkvStride = (long long)TB_D * TB_DH;

  layernorm_f16_kernel<<<TB_TOK, 256, 0, stream>>>(resid_pre, ln1_w, ln1_b, xln);

  wmma_gemm_kernel<<<dim3(32, 1, TB_H), 256, 0, stream>>>(
      xln, TB_D, W_Q, TB_DH, wqkvStride, b_Q, TB_DH, nullptr,
      qb, nullptr, TB_H * TB_DH, TB_DH, TB_D, 0);
  wmma_gemm_kernel<<<dim3(32, 1, TB_H), 256, 0, stream>>>(
      xln, TB_D, W_K, TB_DH, wqkvStride, b_K, TB_DH, nullptr,
      kb, nullptr, TB_H * TB_DH, TB_DH, TB_D, 0);
  wmma_gemm_kernel<<<dim3(32, 1, TB_H), 256, 0, stream>>>(
      xln, TB_D, W_V, TB_DH, wqkvStride, b_V, TB_DH, nullptr,
      vb, nullptr, TB_H * TB_DH, TB_DH, TB_D, 0);

  attn_kernel<<<dim3(TB_S / 64, TB_H, TB_B), 128, 0, stream>>>(qb, kb, vb, zb);

  wmma_gemm_kernel<<<dim3(32, 16, 1), 256, 0, stream>>>(
      zb, TB_H * TB_DH, W_O, TB_D, 0, b_O, 0, resid_pre,
      nullptr, out, TB_D, 0, TB_H * TB_DH, 0);

  layernorm_f16_kernel<<<TB_TOK, 256, 0, stream>>>(out, ln2_w, ln2_b, xln);

  wmma_gemm_kernel<<<dim3(32, TB_DM / 64, 1), 256, 0, stream>>>(
      xln, TB_D, W_in, TB_DM, 0, b_in, 0, nullptr,
      hid, nullptr, TB_DM, 0, TB_D, 1);

  wmma_gemm_kernel<<<dim3(32, 16, 1), 256, 0, stream>>>(
      hid, TB_DM, W_out, TB_D, 0, b_out, 0, out,
      nullptr, out, TB_D, 0, TB_DM, 0);
}
